// SEGRModule_60816736911894
// MI455X (gfx1250) — matrix operand order repaired and checked by reading, not compiled, not run on hardware
//
#include <hip/hip_runtime.h>
#include <hip/hip_bf16.h>
#include <math.h>

#define N_  16
#define C_  256
#define HW_ 4608
#define S_  64
#define EPS_ 1e-5f

typedef __attribute__((ext_vector_type(16))) __bf16 v16bf;
typedef __attribute__((ext_vector_type(8)))  float  v8f;

union FragBF { v16bf v; uint4 q[2]; };

__device__ __forceinline__ unsigned short f2bf(float f) {
    unsigned int u = __float_as_uint(f);
    u += 0x7FFFu + ((u >> 16) & 1u);           // round-to-nearest-even
    return (unsigned short)(u >> 16);
}

// One wave computes a 16x16 f32 tile: D = sum_K A[16xK] * Bt[16xK]^T
// A, Bt: bf16 (ushort) row-major, K-contiguous, 16B-aligned rows.
__device__ __forceinline__ v8f wmma_bf16_tile(const unsigned short* __restrict__ A,
                                              const unsigned short* __restrict__ Bt,
                                              int lda, int ldb, int K, int lane) {
    v8f acc = {0.f, 0.f, 0.f, 0.f, 0.f, 0.f, 0.f, 0.f};
    const int lo = lane & 15;
    const int hi = lane >> 4;
    const unsigned short* pa = A  + (size_t)lo * lda + (hi ? 8  : 0);
    const unsigned short* pb = Bt + (size_t)lo * ldb + (hi ? 8  : 0);
    for (int k = 0; k < K; k += 32) {
        FragBF a, b;
        a.q[0] = *(const uint4*)(pa + k);
        a.q[1] = *(const uint4*)(pa + k + 16);
        b.q[0] = *(const uint4*)(pb + k);
        b.q[1] = *(const uint4*)(pb + k + 16);
        acc = __builtin_amdgcn_wmma_f32_16x16x32_bf16(false, a.v, false, b.v,
                                                      (short)0, acc, false, false);
    }
    return acc;
}

// ---- prep: bf16 weight packs -------------------------------------------------
__global__ void k_prep(const float* __restrict__ Ws, const float* __restrict__ Wp,
                       const float* __restrict__ We,
                       unsigned short* __restrict__ Wsp, unsigned short* __restrict__ We_bf) {
    int idx = blockIdx.x * 256 + threadIdx.x;
    if (idx < 128 * C_) {
        int o = idx >> 8, c = idx & 255;
        float v = (o < S_) ? Ws[o * C_ + c] : Wp[(o - S_) * C_ + c];
        Wsp[idx] = f2bf(v);
    } else if (idx < 128 * C_ + C_ * S_) {
        int j = idx - 128 * C_;
        We_bf[j] = f2bf(We[j]);
    }
}

// ---- edge gate: sigmoid(e1 - e0) --------------------------------------------
__global__ void k_edge(const float* __restrict__ edge, float* __restrict__ eg) {
    int idx = blockIdx.x * 256 + threadIdx.x;
    int n = idx / HW_, p = idx % HW_;
    float e0 = edge[((size_t)n * 2) * HW_ + p];
    float e1 = edge[((size_t)n * 2 + 1) * HW_ + p];
    eg[idx] = 1.0f / (1.0f + __expf(e0 - e1));
}

// ---- x (N,C,HW) f32 -> xT (N,HW,C) bf16, LDS-tiled transpose ----------------
__global__ void k_transpose(const float* __restrict__ x, unsigned short* __restrict__ xT) {
    __shared__ float t[32][33];
    int n = blockIdx.z, p0 = blockIdx.x * 32, c0 = blockIdx.y * 32;
    for (int i = 0; i < 4; ++i) {
        int c = threadIdx.y + 8 * i;
        t[c][threadIdx.x] = x[((size_t)n * C_ + c0 + c) * HW_ + p0 + threadIdx.x];
    }
    __syncthreads();
    for (int i = 0; i < 4; ++i) {
        int pr = threadIdx.y + 8 * i;
        xT[((size_t)n * HW_ + p0 + pr) * C_ + c0 + threadIdx.x] = f2bf(t[threadIdx.x][pr]);
    }
}

// ---- GEMM1: [xs;xp] = x^T * Wsp^T ; epilogue gates xp and stores xs bf16 ----
__global__ void k_gemm1(const unsigned short* __restrict__ xT,
                        const unsigned short* __restrict__ Wsp,
                        const float* __restrict__ bs, const float* __restrict__ bp,
                        const float* __restrict__ eg,
                        unsigned short* __restrict__ xs_bf, float* __restrict__ gated) {
    int wid  = (blockIdx.x * blockDim.x + threadIdx.x) >> 5;
    int lane = threadIdx.x & 31;
    int tn = wid & 7;                 // 128/16 output-channel tiles
    int tm = (wid >> 3) % (HW_ / 16); // spatial tiles
    int n  = wid / (8 * (HW_ / 16));
    const unsigned short* A = xT  + (size_t)n * HW_ * C_ + (size_t)tm * 16 * C_;
    const unsigned short* B = Wsp + (size_t)tn * 16 * C_;
    v8f acc = wmma_bf16_tile(A, B, C_, C_, C_, lane);
    int o  = tn * 16 + (lane & 15);
    int hi = lane >> 4;
    float bias = (o < S_) ? bs[o] : bp[o - S_];
    for (int r = 0; r < 8; ++r) {
        int pos = tm * 16 + r + 8 * hi;
        float v = acc[r] + bias;
        if (o < S_) {
            xs_bf[((size_t)(n * S_ + o)) * HW_ + pos] = f2bf(v);
        } else {
            gated[((size_t)(n * S_ + (o - S_))) * HW_ + pos] =
                v * (1.0f + eg[(size_t)n * HW_ + pos]);
        }
    }
}

// ---- spatial softmax over HW; writes comb (S,HW) and combT (HW,S) bf16 ------
__global__ void k_softmax(const float* __restrict__ gated,
                          unsigned short* __restrict__ comb,
                          unsigned short* __restrict__ combT) {
    __shared__ float red[256];
    int row = blockIdx.x;             // n*S + s
    int tid = threadIdx.x;
    const float* g = gated + (size_t)row * HW_;
    float mx = -3.4e38f;
    for (int p = tid; p < HW_; p += 256) mx = fmaxf(mx, g[p]);
    red[tid] = mx; __syncthreads();
    for (int s = 128; s > 0; s >>= 1) { if (tid < s) red[tid] = fmaxf(red[tid], red[tid + s]); __syncthreads(); }
    mx = red[0]; __syncthreads();
    float sum = 0.f;
    for (int p = tid; p < HW_; p += 256) sum += __expf(g[p] - mx);
    red[tid] = sum; __syncthreads();
    for (int s = 128; s > 0; s >>= 1) { if (tid < s) red[tid] += red[tid + s]; __syncthreads(); }
    float inv = 1.0f / red[0];
    int n = row / S_, sidx = row % S_;
    for (int p = tid; p < HW_; p += 256) {
        unsigned short b = f2bf(__expf(g[p] - mx) * inv);
        comb[(size_t)row * HW_ + p] = b;
        combT[((size_t)n * HW_ + p) * S_ + sidx] = b;
    }
}

// ---- GEMM2: xn[s,t] = sum_k xs[s,k]*comb[t,k], K = HW ------------------------
__global__ void k_gemm2(const unsigned short* __restrict__ xs_bf,
                        const unsigned short* __restrict__ comb,
                        float* __restrict__ xn) {
    int wid  = (blockIdx.x * blockDim.x + threadIdx.x) >> 5;
    int lane = threadIdx.x & 31;
    int tn = wid & 3, tm = (wid >> 2) & 3, n = wid >> 4;
    const unsigned short* A = xs_bf + (size_t)n * S_ * HW_ + (size_t)tm * 16 * HW_;
    const unsigned short* B = comb  + (size_t)n * S_ * HW_ + (size_t)tn * 16 * HW_;
    v8f acc = wmma_bf16_tile(A, B, HW_, HW_, HW_, lane);
    int t = tn * 16 + (lane & 15), hi = lane >> 4;
    for (int r = 0; r < 8; ++r) {
        int s = tm * 16 + r + 8 * hi;
        xn[(size_t)n * S_ * S_ + s * S_ + t] = acc[r];
    }
}

// ---- GCN: hh = relu(W2 @ (W1-mix(xn) - xn)) ; store bf16 ---------------------
__global__ void k_gcn(const float* __restrict__ xn, const float* __restrict__ W1,
                      const float* __restrict__ W2, unsigned short* __restrict__ hh) {
    __shared__ float xs[S_ * S_];
    __shared__ float h1[S_ * S_];
    int n = blockIdx.x, tid = threadIdx.x;
    const float* xnn = xn + (size_t)n * S_ * S_;
    for (int q = tid; q < S_ * S_; q += 256) xs[q] = xnn[q];
    __syncthreads();
    for (int q = tid; q < S_ * S_; q += 256) {
        int s = q >> 6, i = q & 63;
        float acc = 0.f;
        for (int j = 0; j < S_; ++j) acc += W1[i * S_ + j] * xs[s * S_ + j];
        h1[q] = acc - xs[s * S_ + i];
    }
    __syncthreads();
    for (int q = tid; q < S_ * S_; q += 256) {
        int i = q >> 6, k = q & 63;
        float acc = 0.f;
        for (int j = 0; j < S_; ++j) acc += W2[i * S_ + j] * h1[j * S_ + k];
        hh[(size_t)n * S_ * S_ + q] = f2bf(fmaxf(acc, 0.f));
    }
}

// ---- GEMM3: xrT[pos,s] = sum_t combT[pos,t]*hh[s,t] --------------------------
__global__ void k_gemm3(const unsigned short* __restrict__ combT,
                        const unsigned short* __restrict__ hh,
                        unsigned short* __restrict__ xrT) {
    int wid  = (blockIdx.x * blockDim.x + threadIdx.x) >> 5;
    int lane = threadIdx.x & 31;
    int tn = wid & 3;                 // S/16
    int tm = (wid >> 2) % (HW_ / 16);
    int n  = wid / (4 * (HW_ / 16));
    const unsigned short* A = combT + (size_t)n * HW_ * S_ + (size_t)tm * 16 * S_;
    const unsigned short* B = hh    + (size_t)n * S_ * S_  + (size_t)tn * 16 * S_;
    v8f acc = wmma_bf16_tile(A, B, S_, S_, S_, lane);
    int sc = tn * 16 + (lane & 15), hi = lane >> 4;
    for (int r = 0; r < 8; ++r) {
        int pos = tm * 16 + r + 8 * hi;
        xrT[((size_t)n * HW_ + pos) * S_ + sc] = f2bf(acc[r]);
    }
}

// ---- GEMM4: ext[o,pos] = sum_s We[o,s]*xrT[pos,s] (coalesced f32 stores) ----
__global__ void k_gemm4(const unsigned short* __restrict__ We_bf,
                        const unsigned short* __restrict__ xrT,
                        float* __restrict__ ext) {
    int wid  = (blockIdx.x * blockDim.x + threadIdx.x) >> 5;
    int lane = threadIdx.x & 31;
    int tn = wid % (HW_ / 16);
    int tm = (wid / (HW_ / 16)) & 15; // C/16
    int n  = wid / ((HW_ / 16) * 16);
    const unsigned short* A = We_bf + (size_t)tm * 16 * S_;
    const unsigned short* B = xrT + (size_t)n * HW_ * S_ + (size_t)tn * 16 * S_;
    v8f acc = wmma_bf16_tile(A, B, S_, S_, S_, lane);
    int pos = tn * 16 + (lane & 15), hi = lane >> 4;
    for (int r = 0; r < 8; ++r) {
        int o = tm * 16 + r + 8 * hi;
        ext[((size_t)n * C_ + o) * HW_ + pos] = acc[r];
    }
}

// ---- BN batch stats per channel ---------------------------------------------
__global__ void k_bnstats(const float* __restrict__ ext, float* __restrict__ mv) {
    __shared__ float s1[256], s2[256];
    int c = blockIdx.x, tid = threadIdx.x;
    float a = 0.f, b = 0.f;
    for (int idx = tid; idx < N_ * HW_; idx += 256) {
        int n = idx / HW_, p = idx % HW_;
        float v = ext[((size_t)n * C_ + c) * HW_ + p];
        a += v; b += v * v;
    }
    s1[tid] = a; s2[tid] = b; __syncthreads();
    for (int s = 128; s > 0; s >>= 1) {
        if (tid < s) { s1[tid] += s1[tid + s]; s2[tid] += s2[tid + s]; }
        __syncthreads();
    }
    if (tid == 0) {
        float m   = s1[0] / (float)(N_ * HW_);
        float var = s2[0] / (float)(N_ * HW_) - m * m;
        mv[c] = m; mv[C_ + c] = var;
    }
}

// ---- final: out = x + gamma*(ext-mean)*rsqrt(var+eps) + beta ----------------
__global__ void k_final(const float* __restrict__ x, const float* __restrict__ ext,
                        const float* __restrict__ mv, const float* __restrict__ gamma,
                        const float* __restrict__ beta, float* __restrict__ out) {
    size_t idx = (size_t)blockIdx.x * blockDim.x + threadIdx.x;
    int c = (int)((idx / HW_) % C_);
    float m = mv[c], var = mv[C_ + c];
    out[idx] = x[idx] + gamma[c] * (ext[idx] - m) * rsqrtf(var + EPS_) + beta[c];
}

extern "C" void kernel_launch(void* const* d_in, const int* in_sizes, int n_in,
                              void* d_out, int out_size, void* d_ws, size_t ws_size,
                              hipStream_t stream) {
    const float* x     = (const float*)d_in[0];
    const float* edge  = (const float*)d_in[1];
    const float* Ws    = (const float*)d_in[2];
    const float* bs    = (const float*)d_in[3];
    const float* Wp    = (const float*)d_in[4];
    const float* bp    = (const float*)d_in[5];
    const float* W1    = (const float*)d_in[6];
    const float* W2    = (const float*)d_in[7];
    const float* We    = (const float*)d_in[8];
    const float* gamma = (const float*)d_in[9];
    const float* beta  = (const float*)d_in[10];
    float* out = (float*)d_out;

    char* base = (char*)d_ws;
    size_t off = 0;
    auto take = [&](size_t bytes) -> char* {
        char* p = base + off;
        off += (bytes + 255) & ~(size_t)255;
        return p;
    };

    // Region A (75,497,472 B): dead before GEMM4, reused as ext (f32 N*C*HW).
    char* regionA = base;
    unsigned short* xT    = (unsigned short*)take((size_t)N_ * HW_ * C_ * 2);
    float*          gated = (float*)         take((size_t)N_ * S_ * HW_ * 4);
    unsigned short* xs_bf = (unsigned short*)take((size_t)N_ * S_ * HW_ * 2);
    unsigned short* comb  = (unsigned short*)take((size_t)N_ * S_ * HW_ * 2);
    float* ext = (float*)regionA;  // aliases xT|gated|xs_bf|comb exactly

    unsigned short* combT = (unsigned short*)take((size_t)N_ * HW_ * S_ * 2);
    unsigned short* xrT   = (unsigned short*)take((size_t)N_ * HW_ * S_ * 2);
    unsigned short* Wsp   = (unsigned short*)take((size_t)128 * C_ * 2);
    unsigned short* We_bf = (unsigned short*)take((size_t)C_ * S_ * 2);
    float* eg = (float*)take((size_t)N_ * HW_ * 4);
    float* xn = (float*)take((size_t)N_ * S_ * S_ * 4);
    unsigned short* hh = (unsigned short*)take((size_t)N_ * S_ * S_ * 2);
    float* mv = (float*)take((size_t)2 * C_ * 4);

    const int MT = HW_ / 16;  // 288 spatial tiles

    k_prep<<<(128 * C_ + C_ * S_ + 255) / 256, 256, 0, stream>>>(Ws, Wp, We, Wsp, We_bf);
    k_edge<<<(N_ * HW_) / 256, 256, 0, stream>>>(edge, eg);
    k_transpose<<<dim3(HW_ / 32, C_ / 32, N_), dim3(32, 8), 0, stream>>>(x, xT);
    k_gemm1<<<(N_ * MT * 8) / 8, 256, 0, stream>>>(xT, Wsp, bs, bp, eg, xs_bf, gated);
    k_softmax<<<N_ * S_, 256, 0, stream>>>(gated, comb, combT);
    k_gemm2<<<(N_ * 16) / 8, 256, 0, stream>>>(xs_bf, comb, xn);
    k_gcn<<<N_, 256, 0, stream>>>(xn, W1, W2, hh);
    k_gemm3<<<(N_ * MT * 4) / 8, 256, 0, stream>>>(combT, hh, xrT);
    k_gemm4<<<(N_ * 16 * MT) / 8, 256, 0, stream>>>(We_bf, xrT, ext);
    k_bnstats<<<C_, 256, 0, stream>>>(ext, mv);
    k_final<<<(N_ * C_ * HW_) / 256, 256, 0, stream>>>(x, ext, mv, gamma, beta, out);
}
